// CP_HGNN_28166395527441
// MI455X (gfx1250) — matrix operand order repaired and checked by reading, not compiled, not run on hardware
//
#include <hip/hip_runtime.h>
#include <hip/hip_bf16.h>

// ---------------------------------------------------------------------------
// Problem constants (match reference)
// ---------------------------------------------------------------------------
#define NN    40000   // nodes
#define MM    10000   // hyperedges
#define NNZI  400000  // incidence nnz
#define FTD   128
#define HID   256
#define NCLS  10
#define NGR   128
#define BNEPS 1e-5f

typedef __attribute__((ext_vector_type(16))) __bf16 v16bf;
typedef __attribute__((ext_vector_type(8)))  float  v8f;
typedef __attribute__((vector_size(16)))     int    ai4;   // int4 vector for async builtin

// ---------------------------------------------------------------------------
// Helpers
// ---------------------------------------------------------------------------
__device__ __forceinline__ unsigned short f2bf(float x) {
    unsigned u = __float_as_uint(x);
    unsigned r = u + 0x7fffu + ((u >> 16) & 1u);   // round-to-nearest-even
    return (unsigned short)(r >> 16);
}

// Async global->LDS copy of 16 bytes (CDNA5 GLOBAL_LOAD_ASYNC_TO_LDS_B128,
// ASYNCcnt-tracked). Falls back to load + ds_store if builtin unavailable.
__device__ __forceinline__ void async_copy_b128(const unsigned short* g, unsigned short* l) {
#if __has_builtin(__builtin_amdgcn_global_load_async_to_lds_b128)
    __builtin_amdgcn_global_load_async_to_lds_b128(
        (__attribute__((address_space(1))) ai4*)g,
        (__attribute__((address_space(3))) ai4*)l,
        0, 0);
#else
    *(uint4*)l = *(const uint4*)g;
#endif
}

__device__ __forceinline__ void wait_async() {
#if __has_builtin(__builtin_amdgcn_global_load_async_to_lds_b128)
#  if __has_builtin(__builtin_amdgcn_s_wait_asynccnt)
    __builtin_amdgcn_s_wait_asynccnt(0);
#  else
    asm volatile("s_wait_asynccnt 0x0" ::: "memory");
#  endif
#endif
}

__global__ void fill_zero_f32(float* __restrict__ p, size_t n) {
    size_t i = (size_t)blockIdx.x * blockDim.x + threadIdx.x;
    if (i < n) p[i] = 0.0f;
}

// dst[didx[i]*D + c] += src[sidx[i]*D + c]   (D is a power of two: 128 or 256)
template<int LD>
__global__ void scatter_add(const float* __restrict__ src,
                            const int*   __restrict__ sidx,
                            const int*   __restrict__ didx,
                            float*       __restrict__ dst,
                            long long total /* nnz << LD */) {
    long long gid = (long long)blockIdx.x * blockDim.x + threadIdx.x;
    if (gid >= total) return;
    int i = (int)(gid >> LD);
    int c = (int)(gid & ((1 << LD) - 1));
    atomicAdd(&dst[((size_t)didx[i] << LD) + c], src[((size_t)sidx[i] << LD) + c]);
}

__global__ void count_edges(const int* __restrict__ eidx, float* __restrict__ es, int nnz) {
    int i = blockIdx.x * blockDim.x + threadIdx.x;
    if (i < nnz) atomicAdd(&es[eidx[i]], 1.0f);
}

__global__ void rowsum_p(const int* __restrict__ eidx, const int* __restrict__ nidx,
                         const float* __restrict__ es, float* __restrict__ p, int nnz) {
    int i = blockIdx.x * blockDim.x + threadIdx.x;
    if (i < nnz) atomicAdd(&p[nidx[i]], es[eidx[i]]);
}

__global__ void invert_p(float* __restrict__ p, int n) {
    int i = blockIdx.x * blockDim.x + threadIdx.x;
    if (i < n) { float v = p[i]; p[i] = (v > 0.0f) ? 1.0f / v : 1.0f; }
}

__global__ void cvt_bf16(const float* __restrict__ x, unsigned short* __restrict__ y, size_t n) {
    size_t i = (size_t)blockIdx.x * blockDim.x + threadIdx.x;
    if (i < n) y[i] = f2bf(x[i]);
}

// W [Ki x Ko] f32 row-major  ->  Wt [Ko x Ki] bf16 (so B-fragment K-loads are contiguous)
__global__ void wcvt_transpose(const float* __restrict__ W, unsigned short* __restrict__ Wt,
                               int Ki, int Ko) {
    int i = blockIdx.x * blockDim.x + threadIdx.x;
    if (i >= Ki * Ko) return;
    int k = i / Ko, n = i - k * Ko;
    Wt[(size_t)n * Ki + k] = f2bf(W[i]);
}

// ---------------------------------------------------------------------------
// WMMA GEMM:  C[Nrows x 256] = A(bf16)[Nrows x KI] @ Wt(bf16)^T + bias
//
// Block = 256 threads (8 waves). Whole weight matrix Wt [256 x KI] staged in
// LDS once per block (async-to-LDS); per iteration a 32-row A slab is staged.
// Each wave computes a 16x64 strip: one A fragment feeds 4 back-to-back
// v_wmma_f32_16x16x32_bf16. LDS rows padded to KI+8 halfwords (528B pitch ->
// 4-bank stride) so fragment reads are bank-conflict-free.
// Fragment layouts per CDNA5 ISA (16-bit A 16x32, B 32x16, f32 C/D).
// ---------------------------------------------------------------------------
#define MT_PER_BLOCK 4   // 4 row-tiles (64 rows) per block, 2 tiles per iteration

template<int KI>
__global__ void __launch_bounds__(256)
gemm_bf16_wmma(const unsigned short* __restrict__ A,   // [Nrows x KI] bf16 row-major
               const unsigned short* __restrict__ Wt,  // [256 x KI]  bf16 (transposed W)
               const float* __restrict__ bias,         // [256]
               float* __restrict__ C,                  // [Nrows x 256] f32
               int Nrows) {
    const int KO    = HID;
    const int PITCH = KI + 8;                      // halfword pitch (bank decorrelation)
    extern __shared__ char smem[];
    unsigned short* sB = (unsigned short*)smem;                     // [KO][PITCH]
    unsigned short* sA = (unsigned short*)(smem + (size_t)KO * PITCH * 2); // [32][PITCH]

    const int tid  = threadIdx.x;
    const int lane = tid & 31;
    const int wave = tid >> 5;
    const int half = lane & 15;
    const int grp  = lane >> 4;
    const int CPR  = KI / 8;                       // b128 chunks per row

    // ---- stage the full weight matrix into LDS (once per block) ----
    for (int c = tid; c < KO * CPR; c += 256) {
        int n = c / CPR, j = c - n * CPR;
        async_copy_b128(Wt + (size_t)n * KI + j * 8, sB + n * PITCH + j * 8);
    }

#pragma unroll
    for (int it = 0; it < MT_PER_BLOCK / 2; ++it) {
        const int row0 = (blockIdx.x * MT_PER_BLOCK + it * 2) * 16;  // 32-row slab
        const bool live = (row0 < Nrows);                            // block-uniform

        if (live) {   // stage A slab [row0, row0+32)
            for (int c = tid; c < 32 * CPR; c += 256) {
                int r = c / CPR, j = c - r * CPR;
                async_copy_b128(A + (size_t)(row0 + r) * KI + j * 8,
                                sA + r * PITCH + j * 8);
            }
        }
        wait_async();
        __syncthreads();

        if (live) {
            const int mloc = wave >> 2;            // 0..1 : local 16-row tile
            const int ng   = wave & 3;             // 0..3 : 64-col group
            const unsigned short* arow = sA + (size_t)(mloc * 16 + half) * PITCH;

            v8f acc[4] = {};
#pragma unroll
            for (int kb = 0; kb < KI; kb += 32) {
                union { v16bf v; unsigned u[8]; } af;
#pragma unroll
                for (int q = 0; q < 8; ++q) {
                    // A fragment, contiguous k order (same as B): VGPR q holds K={2q,2q+1} (+16*grp)
                    int ka = 2 * q + 16 * grp;
                    af.u[q] = *(const unsigned*)(arow + kb + ka);
                }
#pragma unroll
                for (int t = 0; t < 4; ++t) {      // 4 column tiles reuse one A fragment
                    const unsigned short* brow =
                        sB + (size_t)((ng * 4 + t) * 16 + half) * PITCH + kb + 16 * grp;
                    union { v16bf v; unsigned u[8]; } bf;
#pragma unroll
                    for (int q = 0; q < 8; ++q)
                        bf.u[q] = *(const unsigned*)(brow + 2 * q);
                    acc[t] = __builtin_amdgcn_wmma_f32_16x16x32_bf16(
                        false, af.v, false, bf.v, (short)0, acc[t], false, false);
                }
            }

            const int m0 = row0 + mloc * 16;
#pragma unroll
            for (int t = 0; t < 4; ++t) {
                const int n = (ng * 4 + t) * 16 + half;
                const float b = bias[n];
#pragma unroll
                for (int r = 0; r < 8; ++r)        // C/D: VGPR r -> row m0 + r + 8*grp
                    C[(size_t)(m0 + r + 8 * grp) * KO + n] = acc[t][r] + b;
            }
        }
        __syncthreads();                           // protect sA before next staging
    }
}

// ---------------------------------------------------------------------------
// BatchNorm (training-mode, biased stats) over HID=256 columns
// ---------------------------------------------------------------------------
#define BN_ROWS 128
__global__ void __launch_bounds__(256)
bn_stats(const float* __restrict__ X, float* __restrict__ sums /* [512] sum|sumsq */,
         int Nrows) {
    int c  = threadIdx.x;                   // one column per thread (coalesced rows)
    int r0 = blockIdx.x * BN_ROWS;
    int r1 = r0 + BN_ROWS; if (r1 > Nrows) r1 = Nrows;
    float s = 0.0f, q = 0.0f;
    for (int r = r0; r < r1; ++r) {
        float x = X[(size_t)r * HID + c];
        s += x; q = fmaf(x, x, q);
    }
    atomicAdd(&sums[c], s);
    atomicAdd(&sums[HID + c], q);
}

__global__ void bn_finalize(const float* __restrict__ sums,
                            const float* __restrict__ g, const float* __restrict__ be,
                            float* __restrict__ scale, float* __restrict__ shift,
                            float invN) {
    int c = threadIdx.x;
    float mean = sums[c] * invN;
    float var  = sums[HID + c] * invN - mean * mean;
    float rs   = rsqrtf(var + BNEPS);
    float sc   = g[c] * rs;
    scale[c] = sc;
    shift[c] = fmaf(-mean, sc, be[c]);
}

// y = relu(scale*x + shift); optionally write f32 and/or bf16 outputs
__global__ void bn_apply(const float* __restrict__ X,
                         const float* __restrict__ scale, const float* __restrict__ shift,
                         float* __restrict__ outF, unsigned short* __restrict__ outB,
                         size_t total) {
    size_t i = (size_t)blockIdx.x * blockDim.x + threadIdx.x;
    if (i >= total) return;
    int c = (int)(i & (HID - 1));
    float y = fmaxf(0.0f, fmaf(X[i], scale[c], shift[c]));
    if (outF) outF[i] = y;
    if (outB) outB[i] = f2bf(y);
}

// ---------------------------------------------------------------------------
// Head (512 -> 10) fused with d^-1 row scaling and segment-sum into d_out
// logit[n] = d_inv[n] * (concat(z0,z1)[n] @ Wh) + bh  (linearity lets d_inv hoist)
// ---------------------------------------------------------------------------
__global__ void head_readout(const float* __restrict__ Z0, const float* __restrict__ Z1,
                             const float* __restrict__ dinv,
                             const float* __restrict__ Wh, const float* __restrict__ bh,
                             const int* __restrict__ batch,
                             float* __restrict__ out /* NGR x NCLS */, int Nrows) {
    int n = blockIdx.x * blockDim.x + threadIdx.x;
    if (n >= Nrows) return;
    float acc[NCLS];
#pragma unroll
    for (int c = 0; c < NCLS; ++c) acc[c] = 0.0f;
    const float* z0 = Z0 + (size_t)n * HID;
    const float* z1 = Z1 + (size_t)n * HID;
    for (int k = 0; k < HID; ++k) {
        float x = z0[k];
        const float* w = Wh + (size_t)k * NCLS;
#pragma unroll
        for (int c = 0; c < NCLS; ++c) acc[c] = fmaf(x, w[c], acc[c]);
    }
    for (int k = 0; k < HID; ++k) {
        float x = z1[k];
        const float* w = Wh + (size_t)(HID + k) * NCLS;
#pragma unroll
        for (int c = 0; c < NCLS; ++c) acc[c] = fmaf(x, w[c], acc[c]);
    }
    float s = dinv[n];
    int g = batch[n];
#pragma unroll
    for (int c = 0; c < NCLS; ++c)
        atomicAdd(&out[g * NCLS + c], fmaf(s, acc[c], bh[c]));
}

__global__ void count_batch(const int* __restrict__ b, float* __restrict__ cnt, int n) {
    int i = blockIdx.x * blockDim.x + threadIdx.x;
    if (i < n) atomicAdd(&cnt[b[i]], 1.0f);
}

__global__ void readout_div(float* __restrict__ out, const float* __restrict__ cnt) {
    int i = blockIdx.x * blockDim.x + threadIdx.x;
    if (i < NGR * NCLS) out[i] /= fmaxf(cnt[i / NCLS], 1.0f);
}

// ---------------------------------------------------------------------------
// Host orchestration
// ---------------------------------------------------------------------------
static inline unsigned gblk(long long n, int b) { return (unsigned)((n + b - 1) / b); }

extern "C" void kernel_launch(void* const* d_in, const int* in_sizes, int n_in,
                              void* d_out, int out_size, void* d_ws, size_t ws_size,
                              hipStream_t stream) {
    (void)in_sizes; (void)n_in; (void)out_size; (void)ws_size;

    const float* X        = (const float*)d_in[0];
    const int*   node_idx = (const int*)d_in[1];
    const int*   edge_idx = (const int*)d_in[2];
    const int*   batch    = (const int*)d_in[3];
    const float *W1_0 = (const float*)d_in[4],  *b1_0 = (const float*)d_in[5];
    const float *g1_0 = (const float*)d_in[6],  *be1_0 = (const float*)d_in[7];
    const float *W2_0 = (const float*)d_in[8],  *b2_0 = (const float*)d_in[9];
    const float *bng0 = (const float*)d_in[10], *bnb0 = (const float*)d_in[11];
    const float *W1_1 = (const float*)d_in[12], *b1_1 = (const float*)d_in[13];
    const float *g1_1 = (const float*)d_in[14], *be1_1 = (const float*)d_in[15];
    const float *W2_1 = (const float*)d_in[16], *b2_1 = (const float*)d_in[17];
    const float *bng1 = (const float*)d_in[18], *bnb1 = (const float*)d_in[19];
    const float *headW = (const float*)d_in[20], *headb = (const float*)d_in[21];
    float* out = (float*)d_out;

    // ---- carve workspace (256B aligned) ----
    char* base = (char*)d_ws;
    size_t off = 0;
    auto carve = [&](size_t bytes) -> char* {
        char* p = base + off;
        off += (bytes + 255) & ~(size_t)255;
        return p;
    };
    float*          E     = (float*)carve((size_t)MM * HID * 4);   // edge features
    float*          T     = (float*)carve((size_t)NN * HID * 4);   // sparse out / gemm2 out
    float*          U     = (float*)carve((size_t)NN * HID * 4);   // gemm1 out / h0 normalized
    float*          H0    = (float*)carve((size_t)NN * HID * 4);   // hidden[0]
    float*          H1    = (float*)carve((size_t)NN * HID * 4);   // hidden[1]
    unsigned short* Abf   = (unsigned short*)carve((size_t)NN * HID * 2); // bf16 activations
    unsigned short* Wb    = (unsigned short*)carve((size_t)HID * HID * 2); // bf16 weight (transposed)
    float*          esize = (float*)carve((size_t)MM * 4);
    float*          dinv  = (float*)carve((size_t)NN * 4);
    float*          stats = (float*)carve(1024 * 4);               // sum|sumsq|scale|shift
    float*          cnt   = (float*)carve((size_t)NGR * 4);
    float* scaleP = stats + 512;
    float* shiftP = stats + 768;

    const int B = 256;
    const float invN = 1.0f / (float)NN;

    // ---- degrees: edge_size, p_rowsum, d_inv ----
    fill_zero_f32<<<gblk(MM, B), B, 0, stream>>>(esize, MM);
    fill_zero_f32<<<gblk(NN, B), B, 0, stream>>>(dinv, NN);
    count_edges<<<gblk(NNZI, B), B, 0, stream>>>(edge_idx, esize, NNZI);
    rowsum_p<<<gblk(NNZI, B), B, 0, stream>>>(edge_idx, node_idx, esize, dinv, NNZI);
    invert_p<<<gblk(NN, B), B, 0, stream>>>(dinv, NN);

    // ---- sparse double-propagation helper: dst = H (H^T src), D columns ----
    auto sparse2 = [&](const float* src, float* dst, int ld /*log2 D*/) {
        long long D = 1LL << ld;
        fill_zero_f32<<<gblk((long long)MM * D, B), B, 0, stream>>>(E, (size_t)MM * D);
        fill_zero_f32<<<gblk((long long)NN * D, B), B, 0, stream>>>(dst, (size_t)NN * D);
        if (ld == 7) {
            scatter_add<7><<<gblk((long long)NNZI << 7, B), B, 0, stream>>>(
                src, node_idx, edge_idx, E, (long long)NNZI << 7);
            scatter_add<7><<<gblk((long long)NNZI << 7, B), B, 0, stream>>>(
                E, edge_idx, node_idx, dst, (long long)NNZI << 7);
        } else {
            scatter_add<8><<<gblk((long long)NNZI << 8, B), B, 0, stream>>>(
                src, node_idx, edge_idx, E, (long long)NNZI << 8);
            scatter_add<8><<<gblk((long long)NNZI << 8, B), B, 0, stream>>>(
                E, edge_idx, node_idx, dst, (long long)NNZI << 8);
        }
    };

    const unsigned gemm_blocks = gblk(NN / 16, MT_PER_BLOCK);          // 625
    const size_t shmem128 = (size_t)(HID + 32) * (FTD + 8) * 2;        // B + A slabs
    const size_t shmem256 = (size_t)(HID + 32) * (HID + 8) * 2;
    const size_t nelemH = (size_t)NN * HID;

    // ======================= layer 0 =======================
    sparse2(X, T, 7);                                                   // T = H H^T X  [N,128]
    cvt_bf16<<<gblk((long long)NN * FTD, B), B, 0, stream>>>(T, Abf, (size_t)NN * FTD);
    wcvt_transpose<<<gblk(FTD * HID, B), B, 0, stream>>>(W1_0, Wb, FTD, HID);
    gemm_bf16_wmma<FTD><<<gemm_blocks, 256, shmem128, stream>>>(Abf, Wb, b1_0, U, NN);
    fill_zero_f32<<<2, B, 0, stream>>>(stats, 512);
    bn_stats<<<gblk(NN, BN_ROWS), 256, 0, stream>>>(U, stats, NN);
    bn_finalize<<<1, 256, 0, stream>>>(stats, g1_0, be1_0, scaleP, shiftP, invN);
    bn_apply<<<gblk(nelemH, B), B, 0, stream>>>(U, scaleP, shiftP, nullptr, Abf, nelemH);
    wcvt_transpose<<<gblk(HID * HID, B), B, 0, stream>>>(W2_0, Wb, HID, HID);
    gemm_bf16_wmma<HID><<<gemm_blocks, 256, shmem256, stream>>>(Abf, Wb, b2_0, T, NN);
    fill_zero_f32<<<2, B, 0, stream>>>(stats, 512);
    bn_stats<<<gblk(NN, BN_ROWS), 256, 0, stream>>>(T, stats, NN);
    bn_finalize<<<1, 256, 0, stream>>>(stats, bng0, bnb0, scaleP, shiftP, invN);
    bn_apply<<<gblk(nelemH, B), B, 0, stream>>>(T, scaleP, shiftP, H0, nullptr, nelemH);

    // ======================= layer 1 =======================
    sparse2(H0, T, 8);                                                  // T = H H^T h0 [N,256]
    cvt_bf16<<<gblk(nelemH, B), B, 0, stream>>>(T, Abf, nelemH);
    wcvt_transpose<<<gblk(HID * HID, B), B, 0, stream>>>(W1_1, Wb, HID, HID);
    gemm_bf16_wmma<HID><<<gemm_blocks, 256, shmem256, stream>>>(Abf, Wb, b1_1, U, NN);
    fill_zero_f32<<<2, B, 0, stream>>>(stats, 512);
    bn_stats<<<gblk(NN, BN_ROWS), 256, 0, stream>>>(U, stats, NN);
    bn_finalize<<<1, 256, 0, stream>>>(stats, g1_1, be1_1, scaleP, shiftP, invN);
    bn_apply<<<gblk(nelemH, B), B, 0, stream>>>(U, scaleP, shiftP, nullptr, Abf, nelemH);
    wcvt_transpose<<<gblk(HID * HID, B), B, 0, stream>>>(W2_1, Wb, HID, HID);
    gemm_bf16_wmma<HID><<<gemm_blocks, 256, shmem256, stream>>>(Abf, Wb, b2_1, T, NN);
    fill_zero_f32<<<2, B, 0, stream>>>(stats, 512);
    bn_stats<<<gblk(NN, BN_ROWS), 256, 0, stream>>>(T, stats, NN);
    bn_finalize<<<1, 256, 0, stream>>>(stats, bng1, bnb1, scaleP, shiftP, invN);
    bn_apply<<<gblk(nelemH, B), B, 0, stream>>>(T, scaleP, shiftP, H1, nullptr, nelemH);

    // ---- mean-readout normalization pass (d_inv fused into head) ----
    sparse2(H0, U, 8);                                                  // U = H H^T h0
    sparse2(H1, T, 8);                                                  // T = H H^T h1

    // ---- head + per-graph segment mean ----
    fill_zero_f32<<<gblk(NGR * NCLS, B), B, 0, stream>>>(out, (size_t)NGR * NCLS);
    fill_zero_f32<<<1, B, 0, stream>>>(cnt, NGR);
    count_batch<<<gblk(NN, B), B, 0, stream>>>(batch, cnt, NN);
    head_readout<<<gblk(NN, B), B, 0, stream>>>(U, T, dinv, headW, headb, batch, out, NN);
    readout_div<<<gblk(NGR * NCLS, B), B, 0, stream>>>(out, cnt);
}
